// MultiHeadAttention_81157702025385
// MI455X (gfx1250) — matrix operand order repaired and checked by reading, not compiled, not run on hardware
//
#include <hip/hip_runtime.h>
#include <hip/hip_bf16.h>

// ---------------------------------------------------------------------------
// MHA forward for MI455X (gfx1250): bf16 WMMA GEMMs + flash-style attention,
// with async global->LDS tile movement (ASYNCcnt) and double buffering.
// B=4, S=2048, DIN=DOUT=2048, H=16, D=128.  M = B*S = 8192.
// ---------------------------------------------------------------------------

typedef __attribute__((ext_vector_type(16))) __bf16        v16bf;
typedef __attribute__((ext_vector_type(8)))  float         v8f;
typedef __attribute__((ext_vector_type(4)))  unsigned int  u32x4;

#define MHA_B    4
#define MHA_S    2048
#define MHA_H    16
#define MHA_D    128
#define MHA_M    (MHA_B * MHA_S)   // 8192
#define MHA_N    2048
#define MHA_K    2048
#define SM_SCALE 0.022097086912079608f   // 1/sqrt(2048)

union FragU { v16bf v; u32x4 q[2]; };

// ---------------------------------------------------------------------------
// CDNA5 async global->LDS (tracked by ASYNCcnt).  VDST VGPR holds the LDS
// byte address; address is a 64-bit VGPR
// pair; "off" = no SADDR.
// ---------------------------------------------------------------------------
__device__ __forceinline__ void async_copy_b128(void* lds_dst, const void* gsrc) {
    unsigned int l = (unsigned int)(unsigned long long)lds_dst;
    asm volatile("global_load_async_to_lds_b128 %0, %1, off"
                 :: "v"(l), "v"((unsigned long long)gsrc) : "memory");
}
__device__ __forceinline__ void wait_async() {
    asm volatile("s_wait_asynccnt 0x0" ::: "memory");
}

// A-operand fragment (16x32 bf16), per CDNA5 ISA layout:
// lane r (0..15): row r; half hi: VGPR0-3 hold K=hi*8..+7, VGPR4-7 K=hi*8+16..+23
__device__ __forceinline__ v16bf frag_a(const __bf16* base, int ld, int lane) {
    int r  = lane & 15;
    int hi = lane >> 4;
    const __bf16* p = base + r * ld + hi * 8;
    FragU f;
    f.q[0] = *(const u32x4*)(p);
    f.q[1] = *(const u32x4*)(p + 16);
    return f.v;
}

// B-operand fragment (32x16 bf16) from an N-major (transposed) tile:
// lane n (0..15): column n; half hi: VGPR0-3 hold K=hi*8..+7, VGPR4-7 K=hi*8+16..+23
__device__ __forceinline__ v16bf frag_b(const __bf16* base, int ld, int lane) {
    int n  = lane & 15;
    int hi = lane >> 4;
    const __bf16* p = base + n * ld + hi * 8;
    FragU f;
    f.q[0] = *(const u32x4*)(p);
    f.q[1] = *(const u32x4*)(p + 16);
    return f.v;
}

__device__ __forceinline__ v8f wmma_bf16(v16bf a, v16bf b, v8f c) {
    return __builtin_amdgcn_wmma_f32_16x16x32_bf16(false, a, false, b, (short)0, c, false, false);
}

__device__ __forceinline__ v8f v8f_zero() {
    v8f z = {0.f, 0.f, 0.f, 0.f, 0.f, 0.f, 0.f, 0.f};
    return z;
}

__device__ __forceinline__ void store_out(float* p, unsigned int i, float v)  { p[i] = v; }
__device__ __forceinline__ void store_out(__bf16* p, unsigned int i, float v) { p[i] = (__bf16)v; }

// ---------------------------------------------------------------------------
// fp32 -> bf16 cast
// ---------------------------------------------------------------------------
__global__ __launch_bounds__(256) void cast_f32_bf16(const float* __restrict__ in,
                                                     __bf16* __restrict__ out,
                                                     unsigned int n) {
    unsigned int i      = blockIdx.x * 256u + threadIdx.x;
    unsigned int stride = gridDim.x * 256u;
    for (; i < n; i += stride) out[i] = (__bf16)in[i];
}

// ---------------------------------------------------------------------------
// Tiled bf16 GEMM:  C[M,N] = A[M,K] * W[K,N]
// 256x128 block tile, 8 waves in 4(M)x2(N), each owning 64x64 (16 accums).
// K-step 32; double-buffered LDS; async A-tile loads; W tile transposed in
// LDS via coalesced b128 + scatter.  16 WMMA per 16 ds_load_b128 per K-step.
// ---------------------------------------------------------------------------
template <typename OUT>
__global__ __launch_bounds__(256) void gemm_bf16(const __bf16* __restrict__ A,
                                                 const __bf16* __restrict__ W,
                                                 OUT* __restrict__ out,
                                                 int M, int N, int K) {
    __shared__ __bf16 As[2][256][40];   // row-major A tile, padded (80B stride)
    __shared__ __bf16 Bt[2][128][40];   // W tile stored transposed: Bt[n][k]

    const int tid  = threadIdx.x;
    const int lane = tid & 31;
    const int w    = tid >> 5;
    const int half = lane >> 4;
    const int lcol = lane & 15;
    const int wm   = w >> 1;            // 0..3 : 64-row slab
    const int wn   = w & 1;             // 0..1 : 64-col slab
    const unsigned int m0 = blockIdx.y * 256u;
    const unsigned int n0 = blockIdx.x * 128u;
    const unsigned int uN = (unsigned int)N;
    const unsigned int uK = (unsigned int)K;

    auto load_a = [&](int buf, unsigned int k0) {
#pragma unroll
        for (int cc = 0; cc < 4; ++cc) {           // 1024 x 16B chunks
            int c   = tid + cc * 256;
            int row = c >> 2;                      // 0..255
            int off = (c & 3) * 8;
            async_copy_b128(&As[buf][row][off],
                            &A[(m0 + row) * uK + k0 + off]);
        }
    };
    auto load_b = [&](int buf, unsigned int k0) {
#pragma unroll
        for (int cc = 0; cc < 2; ++cc) {           // 512 x 16B chunks
            int c  = tid + cc * 256;
            int kr = c >> 4;                       // 0..31
            int n8 = (c & 15) * 8;
            u32x4 t = *(const u32x4*)&W[(k0 + kr) * uN + n0 + n8];
            const __bf16* tp = (const __bf16*)&t;
#pragma unroll
            for (int j = 0; j < 8; ++j) Bt[buf][n8 + j][kr] = tp[j];
        }
    };

    v8f acc[4][4];
#pragma unroll
    for (int i = 0; i < 4; ++i)
#pragma unroll
        for (int j = 0; j < 4; ++j) acc[i][j] = v8f_zero();

    load_a(0, 0);
    load_b(0, 0);

    const int nk = K / 32;
    for (int t = 0; t < nk; ++t) {
        wait_async();
        __syncthreads();               // tile t fully in LDS, prev reads done
        const int buf = t & 1;
        if (t + 1 < nk) {              // prefetch tile t+1 into other buffer
            load_a(buf ^ 1, (unsigned int)(t + 1) * 32u);
            load_b(buf ^ 1, (unsigned int)(t + 1) * 32u);
        }
        v16bf af[4];
#pragma unroll
        for (int i = 0; i < 4; ++i)
            af[i] = frag_a(&As[buf][wm * 64 + i * 16][0], 40, lane);
#pragma unroll
        for (int j = 0; j < 4; ++j) {
            v16bf bf = frag_b(&Bt[buf][wn * 64 + j * 16][0], 40, lane);
#pragma unroll
            for (int i = 0; i < 4; ++i)
                acc[i][j] = wmma_bf16(af[i], bf, acc[i][j]);
        }
    }

    // Epilogue: C layout = lane holds column lcol, VGPR v holds row v + 8*half.
#pragma unroll
    for (int i = 0; i < 4; ++i)
#pragma unroll
        for (int j = 0; j < 4; ++j) {
            unsigned int row  = m0 + wm * 64 + i * 16 + half * 8;
            unsigned int col  = n0 + wn * 64 + j * 16 + lcol;
            unsigned int base = row * uN + col;
#pragma unroll
            for (int v = 0; v < 8; ++v)
                store_out(out, base + (unsigned int)v * uN, acc[i][j][v]);
        }
}

// ---------------------------------------------------------------------------
// RoPE (in place, Q and K)
// ---------------------------------------------------------------------------
__global__ __launch_bounds__(256) void rope_kernel(__bf16* __restrict__ Q,
                                                   __bf16* __restrict__ Kb,
                                                   const float* __restrict__ cosT,
                                                   const float* __restrict__ sinT) {
    unsigned int idx = blockIdx.x * 256u + threadIdx.x;   // one per (row, head, d<64)
    if (idx >= (unsigned int)(MHA_M * MHA_H * (MHA_D / 2))) return;
    int d = idx & 63;
    int h = (idx >> 6) & 15;
    unsigned int g = idx >> 10;                 // global row in [0, B*S)
    int s = g & (MHA_S - 1);
    float c1 = cosT[s * MHA_D + d];
    float s1 = sinT[s * MHA_D + d];
    float c2 = cosT[s * MHA_D + d + 64];
    float s2 = sinT[s * MHA_D + d + 64];
    unsigned int base = g * (unsigned int)MHA_N + h * MHA_D + d;

    float q1 = (float)Q[base], q2 = (float)Q[base + 64];
    Q[base]      = (__bf16)(q1 * c1 - q2 * s1);
    Q[base + 64] = (__bf16)(q2 * c2 + q1 * s2);

    float k1 = (float)Kb[base], k2 = (float)Kb[base + 64];
    Kb[base]      = (__bf16)(k1 * c1 - k2 * s1);
    Kb[base + 64] = (__bf16)(k2 * c2 + k1 * s2);
}

// ---------------------------------------------------------------------------
// Flash-style attention with "zero-masked-score" softmax (matches reference:
// masked scores are 0, NOT -inf, and participate in the softmax).
// One block per (q-tile of 128, head, batch).  8 waves; wave w owns q-rows
// [w*16, w*16+16) so online-softmax row reductions stay inside the wave.
// ---------------------------------------------------------------------------
__global__ __launch_bounds__(256) void attention_kernel(const __bf16* __restrict__ Qg,
                                                        const __bf16* __restrict__ Kg,
                                                        const __bf16* __restrict__ Vg,
                                                        __bf16* __restrict__ Og) {
    __shared__ __bf16 Qs[128][136];        // [q_row][d]
    __shared__ __bf16 Ks[128][136];        // [k_row][d]  (== B-operand layout for Q*K^T)
    __shared__ __bf16 Vt[128][136];        // [d][k_row]  (B-operand layout for P*V)
    __shared__ __bf16 Ps[8][16][136];      // per-wave P tile [q_row][k_row]

    const int tid  = threadIdx.x;
    const int lane = tid & 31;
    const int w    = tid >> 5;
    const int half = lane >> 4;
    const int lcol = lane & 15;
    const int qt = blockIdx.x;
    const int h  = blockIdx.y;
    const int b  = blockIdx.z;

    const unsigned int gq0  = (unsigned int)b * MHA_S + qt * 128u;
    const unsigned int col0 = h * MHA_D;

    // Load Q tile (128 x 128 bf16) asynchronously: 2048 x 16B chunks.
#pragma unroll
    for (int i = 0; i < 8; ++i) {
        int c = tid + i * 256;
        int r = c >> 4;
        int o = (c & 15) * 8;
        async_copy_b128(&Qs[r][o], &Qg[(gq0 + r) * (unsigned int)MHA_N + col0 + o]);
    }

    float m_r[8], l_r[8];
    v8f oacc[8];
#pragma unroll
    for (int v = 0; v < 8; ++v) { m_r[v] = -1e30f; l_r[v] = 0.f; }
#pragma unroll
    for (int a = 0; a < 8; ++a) oacc[a] = v8f_zero();

    const int qbase = qt * 128 + w * 16;

    for (int kt = 0; kt < MHA_S / 128; ++kt) {
        const unsigned int gk0 = (unsigned int)b * MHA_S + kt * 128u;
        __syncthreads();   // previous iteration's reads of Ks/Vt are done
        // K tile via async-to-LDS; V tile transposed via b128 + scatter.
#pragma unroll
        for (int i = 0; i < 8; ++i) {
            int c = tid + i * 256;
            int r = c >> 4;
            int o = (c & 15) * 8;
            async_copy_b128(&Ks[r][o], &Kg[(gk0 + r) * (unsigned int)MHA_N + col0 + o]);
            u32x4 tv = *(const u32x4*)&Vg[(gk0 + r) * (unsigned int)MHA_N + col0 + o];
            const __bf16* tp = (const __bf16*)&tv;
#pragma unroll
            for (int j = 0; j < 8; ++j) Vt[o + j][r] = tp[j];
        }
        wait_async();
        __syncthreads();

        // ---- S = Q * K^T  (M=16 q-rows, N=128 k-cols, contraction over d=128)
        v8f sc[8];
#pragma unroll
        for (int a = 0; a < 8; ++a) sc[a] = v8f_zero();
#pragma unroll
        for (int ks = 0; ks < 4; ++ks) {
            v16bf af = frag_a(&Qs[w * 16][ks * 32], 136, lane);
#pragma unroll
            for (int a = 0; a < 8; ++a) {
                v16bf bf = frag_b(&Ks[a * 16][ks * 32], 136, lane);
                sc[a] = wmma_bf16(af, bf, sc[a]);
            }
        }

        // ---- scale + causal zero-mask (reference multiplies scores by tril mask)
#pragma unroll
        for (int a = 0; a < 8; ++a)
#pragma unroll
            for (int v = 0; v < 8; ++v) {
                int qi = qbase + v + 8 * half;
                int ki = kt * 128 + a * 16 + lcol;
                float x = sc[a][v];
                sc[a][v] = (ki <= qi) ? x * SM_SCALE : 0.0f;
            }

        // ---- online softmax (rows live in (VGPR v, lane-half); reduce 16 lanes)
        float m_new[8], csc[8];
#pragma unroll
        for (int v = 0; v < 8; ++v) {
            float mx = sc[0][v];
#pragma unroll
            for (int a = 1; a < 8; ++a) mx = fmaxf(mx, sc[a][v]);
#pragma unroll
            for (int off = 1; off <= 8; off <<= 1)
                mx = fmaxf(mx, __shfl_xor(mx, off, 32));
            m_new[v] = fmaxf(m_r[v], mx);
            csc[v]   = __expf(m_r[v] - m_new[v]);
        }
#pragma unroll
        for (int a = 0; a < 8; ++a)
#pragma unroll
            for (int v = 0; v < 8; ++v)
                sc[a][v] = __expf(sc[a][v] - m_new[v]);
#pragma unroll
        for (int v = 0; v < 8; ++v) {
            float sm = 0.f;
#pragma unroll
            for (int a = 0; a < 8; ++a) sm += sc[a][v];
#pragma unroll
            for (int off = 1; off <= 8; off <<= 1)
                sm += __shfl_xor(sm, off, 32);
            l_r[v] = l_r[v] * csc[v] + sm;
            m_r[v] = m_new[v];
        }
#pragma unroll
        for (int a = 0; a < 8; ++a)
#pragma unroll
            for (int v = 0; v < 8; ++v) oacc[a][v] *= csc[v];

        // ---- write P (C-layout -> LDS) so we can re-read it in A-operand layout
#pragma unroll
        for (int a = 0; a < 8; ++a)
#pragma unroll
            for (int v = 0; v < 8; ++v)
                Ps[w][v + 8 * half][a * 16 + lcol] = (__bf16)sc[a][v];
        __syncthreads();

        // ---- O += P * V   (contraction over 128 k-rows; B operand from Vt)
#pragma unroll
        for (int ks = 0; ks < 4; ++ks) {
            v16bf pf = frag_a(&Ps[w][0][ks * 32], 136, lane);
#pragma unroll
            for (int a = 0; a < 8; ++a) {
                v16bf vf = frag_b(&Vt[a * 16][ks * 32], 136, lane);
                oacc[a] = wmma_bf16(pf, vf, oacc[a]);
            }
        }
    }

    // ---- epilogue: divide by l, store bf16 to O[b, s, h*128 + d]
#pragma unroll
    for (int v = 0; v < 8; ++v) {
        float inv = 1.0f / l_r[v];
        unsigned int g = gq0 + w * 16 + v + 8 * half;
#pragma unroll
        for (int a = 0; a < 8; ++a) {
            unsigned int col = col0 + a * 16 + lcol;
            Og[g * (unsigned int)MHA_N + col] = (__bf16)(oacc[a][v] * inv);
        }
    }
}

// ---------------------------------------------------------------------------
// Launch
// ---------------------------------------------------------------------------
extern "C" void kernel_launch(void* const* d_in, const int* in_sizes, int n_in,
                              void* d_out, int out_size, void* d_ws, size_t ws_size,
                              hipStream_t stream) {
    const float* x    = (const float*)d_in[0];
    const float* Wq   = (const float*)d_in[1];
    const float* Wk   = (const float*)d_in[2];
    const float* Wv   = (const float*)d_in[3];
    const float* Wo   = (const float*)d_in[4];
    const float* cosT = (const float*)d_in[5];
    const float* sinT = (const float*)d_in[6];
    float* out = (float*)d_out;

    const size_t NX = (size_t)MHA_M * MHA_K;   // 16,777,216
    const size_t NW = (size_t)MHA_K * MHA_N;   //  4,194,304

    char* ws = (char*)d_ws;
    __bf16* xb  = (__bf16*)ws;                       // 32 MiB
    __bf16* wqb = (__bf16*)(ws + NX * 2);
    __bf16* wkb = wqb + NW;
    __bf16* wvb = wkb + NW;
    __bf16* wob = wvb + NW;
    __bf16* Qb  = wob + NW;                          // 32 MiB each
    __bf16* Kb  = Qb + NX;
    __bf16* Vb  = Kb + NX;
    __bf16* Ob  = xb;                                // reuse xb after QKV GEMMs

    // 1) casts
    cast_f32_bf16<<<4096, 256, 0, stream>>>(x,  xb,  (unsigned int)NX);
    cast_f32_bf16<<<2048, 256, 0, stream>>>(Wq, wqb, (unsigned int)NW);
    cast_f32_bf16<<<2048, 256, 0, stream>>>(Wk, wkb, (unsigned int)NW);
    cast_f32_bf16<<<2048, 256, 0, stream>>>(Wv, wvb, (unsigned int)NW);
    cast_f32_bf16<<<2048, 256, 0, stream>>>(Wo, wob, (unsigned int)NW);

    // 2) QKV projections (bf16 out)
    dim3 gg(MHA_N / 128, MHA_M / 256);   // (16, 32)
    gemm_bf16<__bf16><<<gg, 256, 0, stream>>>(xb, wqb, Qb, MHA_M, MHA_N, MHA_K);
    gemm_bf16<__bf16><<<gg, 256, 0, stream>>>(xb, wkb, Kb, MHA_M, MHA_N, MHA_K);
    gemm_bf16<__bf16><<<gg, 256, 0, stream>>>(xb, wvb, Vb, MHA_M, MHA_N, MHA_K);

    // 3) RoPE on Q and K (in place)
    rope_kernel<<<(MHA_M * MHA_H * (MHA_D / 2)) / 256, 256, 0, stream>>>(Qb, Kb, cosT, sinT);

    // 4) attention -> Ob (reuses xb region; xb no longer needed)
    dim3 ga(MHA_S / 128, MHA_H, MHA_B);  // (16, 16, 4)
    attention_kernel<<<ga, 256, 0, stream>>>(Qb, Kb, Vb, Ob);

    // 5) output projection -> fp32 d_out
    gemm_bf16<float><<<gg, 256, 0, stream>>>(Ob, wob, out, MHA_M, MHA_N, MHA_K);
}
